// AttentionHeadv2_18262200942725
// MI455X (gfx1250) — matrix operand order repaired and checked by reading, not compiled, not run on hardware
//
#include <hip/hip_runtime.h>
#include <cmath>

typedef __bf16 bf16_t;
typedef __attribute__((ext_vector_type(8)))  __bf16 v8bf;
typedef __attribute__((ext_vector_type(16))) __bf16 v16bf;
typedef __attribute__((ext_vector_type(8)))  float  v8f;
typedef __attribute__((ext_vector_type(4)))  float  v4f;

#define Nn 4096
#define Cc 256
#define HW 49
#define Ss 64     // spatial padded to 4 N-tiles
#define KCn 128
#define VCn 256

__device__ __forceinline__ v16bf cat16(v8bf lo, v8bf hi) {
  return __builtin_shufflevector(lo, hi, 0,1,2,3,4,5,6,7,8,9,10,11,12,13,14,15);
}

// ---- WMMA fragment loaders (CDNA5 ISA layouts), all 16B vector accesses ----
// A fragment 16x32 bf16 from row-major [M][K] (lda elems), tile origin (r0,k0).
// lanes 0-15 hold K 0-7 / 16-23 ; lanes 16-31 hold K 8-15 / 24-31
__device__ __forceinline__ v16bf load_a_rm(const bf16_t* A, int lda, int r0, int k0, int lane) {
  int r   = lane & 15;
  int off = (lane >> 4) << 3;
  const bf16_t* p = A + (size_t)(r0 + r) * lda + k0 + off;
  return cat16(*(const v8bf*)p, *(const v8bf*)(p + 16));
}

// B fragment 32x16 bf16 where the source is stored [N][K] (ldm elems): contiguous in K.
// lane holds column n ; lanes 0-15: K 0-7 / 16-23, lanes 16-31: K 8-15 / 24-31
__device__ __forceinline__ v16bf load_b_cm(const bf16_t* M, int ldm, int k0, int n0, int lane) {
  int n  = lane & 15;
  int kk = k0 + ((lane >> 4) << 3);
  const bf16_t* p = M + (size_t)(n0 + n) * ldm + kk;
  return cat16(*(const v8bf*)p, *(const v8bf*)(p + 16));
}

#define WMMA_BF16(acc, a, b) \
  acc = __builtin_amdgcn_wmma_f32_16x16x32_bf16(false, (a), false, (b), (short)0, (acc), false, false)

// ---------------- weight fp32 -> bf16 pre-pass ----------------
__global__ void cvt_weights(const float* Wk, const float* Wq, const float* Wv, const float* Wr,
                            bf16_t* wk, bf16_t* wq, bf16_t* wv, bf16_t* wr) {
  int i = blockIdx.x * 256 + threadIdx.x;
  if (i < KCn * Cc) { wk[i] = (bf16_t)Wk[i]; wq[i] = (bf16_t)Wq[i]; }
  if (i < VCn * Cc) { wv[i] = (bf16_t)Wv[i]; wr[i] = (bf16_t)Wr[i]; }
}

// ---------------- fused linear-attention head, one block per sample ----------------
__global__ void __launch_bounds__(256) attn_head_kernel(
    const float* __restrict__ x,
    const float* __restrict__ bk, const float* __restrict__ bq, const float* __restrict__ bv,
    const float* __restrict__ br, const float* __restrict__ Wl, const float* __restrict__ bl,
    const bf16_t* __restrict__ wk, const bf16_t* __restrict__ wq,
    const bf16_t* __restrict__ wv, const bf16_t* __restrict__ wr,
    float* __restrict__ pred, float* __restrict__ feats)
{
  extern __shared__ char smem[];
  // A [0,32K):    sXT   bf16 [64][256]  (x transposed) ; reused as sAttT after KQV phase
  // B [32K,48K):  sK    bf16 [128][64]  (row-major, A operand of ctx GEMM)
  // C [48K,64K):  sQT   bf16 [64][128]  (queries transposed, B operand)
  // D [64K,96K):  sV    bf16 [256][64]  (row-major; B^T operand of ctx GEMM)
  // E [96K,160K): sCtxT bf16 [256][128] (context transposed, A operand of att GEMM)
  // B..D reused as sRT f32 [64][256] during reprojection
  bf16_t* sXT   = (bf16_t*)(smem);
  bf16_t* sK    = (bf16_t*)(smem + 32768);
  bf16_t* sQT   = (bf16_t*)(smem + 49152);
  bf16_t* sV    = (bf16_t*)(smem + 65536);
  bf16_t* sCtxT = (bf16_t*)(smem + 98304);
  bf16_t* sAttT = sXT;
  float*  sRT   = (float*)(smem + 32768);

  const int n    = blockIdx.x;
  const int tid  = threadIdx.x;
  const int lane = tid & 31;
  const int wave = tid >> 5;
  const float* xn = x + (size_t)n * Cc * HW;

  // ---- stage x -> LDS transposed bf16 [s][c], zero spatial padding (s 49..63) ----
  for (int i = tid; i < Cc * Ss; i += 256) {
    int c = i >> 6, s = i & 63;                       // coalesced global read over s
    sXT[s * Cc + c] = (bf16_t)((s < HW) ? xn[c * HW + s] : 0.0f);
  }
  __syncthreads();

  // ---- phase 2: K, Q, V GEMMs fused: 32 M-strips (8 K + 8 Q + 16 V), A held in VGPRs ----
  for (int st = wave; st < 32; st += 8) {
    const bf16_t* W; const float* bias; int mt, kind;
    if (st < 8)       { kind = 0; W = wk; bias = bk; mt = st; }
    else if (st < 16) { kind = 1; W = wq; bias = bq; mt = st - 8; }
    else              { kind = 2; W = wv; bias = bv; mt = st - 16; }
    v16bf a[8];
#pragma unroll
    for (int kk = 0; kk < 8; ++kk) a[kk] = load_a_rm(W, Cc, mt * 16, kk * 32, lane);
    int rb = mt * 16 + ((lane >> 4) << 3);            // channel base (multiple of 8)
    v4f bb0 = *(const v4f*)(bias + rb);
    v4f bb1 = *(const v4f*)(bias + rb + 4);
#pragma unroll
    for (int nt = 0; nt < 4; ++nt) {                  // reuse A strip over all spatial tiles
      v8f acc = {};
#pragma unroll
      for (int kk = 0; kk < 8; ++kk)
        WMMA_BF16(acc, a[kk], load_b_cm(sXT, Cc, kk * 32, nt * 16, lane));
      int col = nt * 16 + (lane & 15);                // spatial index
      if (kind == 1) {                                // queries -> sQT[s][kc], contiguous store
        v8bf o;
#pragma unroll
        for (int j = 0; j < 8; ++j) {
          float bj = (j < 4) ? bb0[j] : bb1[j - 4];
          o[j] = (bf16_t)fmaxf(acc[j] + bj, 0.0f);
        }
        *(v8bf*)(&sQT[col * KCn + rb]) = o;
      } else {                                        // keys/values row-major
        bf16_t* dst = (kind == 0) ? sK : sV;
#pragma unroll
        for (int j = 0; j < 8; ++j) {
          float bj = (j < 4) ? bb0[j] : bb1[j - 4];
          dst[(rb + j) * Ss + col] = (bf16_t)fmaxf(acc[j] + bj, 0.0f);
        }
      }
    }
  }
  __syncthreads();

  // ---- phase 3: both softmaxes ----
  if (tid < KCn) {                                    // key softmax over spatial (49), zero pad
    float m = -1e30f;
    for (int s = 0; s < HW; ++s) m = fmaxf(m, (float)sK[tid * Ss + s]);
    float sum = 0.0f;
    for (int s = 0; s < HW; ++s) sum += __expf((float)sK[tid * Ss + s] - m);
    float inv = 1.0f / sum;
    for (int s = 0; s < HW; ++s)
      sK[tid * Ss + s] = (bf16_t)(__expf((float)sK[tid * Ss + s] - m) * inv);
    for (int s = HW; s < Ss; ++s) sK[tid * Ss + s] = (bf16_t)0.0f;
  } else if (tid < KCn + Ss) {                        // query softmax over channels (contig row)
    int s = tid - KCn;
    float m = -1e30f;
    for (int r = 0; r < KCn; ++r) m = fmaxf(m, (float)sQT[s * KCn + r]);
    float sum = 0.0f;
    for (int r = 0; r < KCn; ++r) sum += __expf((float)sQT[s * KCn + r] - m);
    float inv = 1.0f / sum;
    for (int r = 0; r < KCn; ++r)
      sQT[s * KCn + r] = (bf16_t)(__expf((float)sQT[s * KCn + r] - m) * inv);
  }
  __syncthreads();

  // ---- phase 4: context = key_sm @ V^T -> sCtxT [v][kc]; one KC strip per wave ----
  {
    int mt = wave;                                    // 8 strips over KC, exactly 8 waves
    v16bf a0 = load_a_rm(sK, Ss, mt * 16, 0, lane);
    v16bf a1 = load_a_rm(sK, Ss, mt * 16, 32, lane);
    int rb = mt * 16 + ((lane >> 4) << 3);            // kc base
#pragma unroll
    for (int nt = 0; nt < 16; ++nt) {                 // all VC tiles reuse the A strip
      v8f acc = {};
      WMMA_BF16(acc, a0, load_b_cm(sV, Ss, 0, nt * 16, lane));
      WMMA_BF16(acc, a1, load_b_cm(sV, Ss, 32, nt * 16, lane));
      int col = nt * 16 + (lane & 15);                // v index
      v8bf o;
#pragma unroll
      for (int j = 0; j < 8; ++j) o[j] = (bf16_t)acc[j];
      *(v8bf*)(&sCtxT[col * KCn + rb]) = o;           // transposed store, contiguous
    }
  }
  __syncthreads();

  // ---- phase 5: attended = context^T @ query_sm -> sAttT [s][v] (overwrites sXT) ----
  for (int st = wave; st < 16; st += 8) {             // 16 VC strips, 2 per wave
    v16bf a[4];
#pragma unroll
    for (int kk = 0; kk < 4; ++kk) a[kk] = load_a_rm(sCtxT, KCn, st * 16, kk * 32, lane);
    int rb = st * 16 + ((lane >> 4) << 3);            // v base
#pragma unroll
    for (int nt = 0; nt < 4; ++nt) {
      v8f acc = {};
#pragma unroll
      for (int kk = 0; kk < 4; ++kk)
        WMMA_BF16(acc, a[kk], load_b_cm(sQT, KCn, kk * 32, nt * 16, lane));
      int col = nt * 16 + (lane & 15);                // s index
      v8bf o;
#pragma unroll
      for (int j = 0; j < 8; ++j) o[j] = (bf16_t)acc[j];
      *(v8bf*)(&sAttT[col * VCn + rb]) = o;           // [s][v], contiguous
    }
  }
  __syncthreads();

  // ---- phase 6: reproj = relu(Wr @ attended + br); feats = reproj + x; stash reproj f32 ----
  for (int st = wave; st < 16; st += 8) {             // 16 C strips, 2 per wave
    v16bf a[8];
#pragma unroll
    for (int kk = 0; kk < 8; ++kk) a[kk] = load_a_rm(wr, VCn, st * 16, kk * 32, lane);
    int rb = st * 16 + ((lane >> 4) << 3);            // c base
    v4f bb0 = *(const v4f*)(br + rb);
    v4f bb1 = *(const v4f*)(br + rb + 4);
#pragma unroll
    for (int nt = 0; nt < 4; ++nt) {
      v8f acc = {};
#pragma unroll
      for (int kk = 0; kk < 8; ++kk)
        WMMA_BF16(acc, a[kk], load_b_cm(sAttT, VCn, kk * 32, nt * 16, lane));
      int col = nt * 16 + (lane & 15);                // s index
      v4f o0, o1;
#pragma unroll
      for (int j = 0; j < 8; ++j) {
        int row = rb + j;
        float bj = (j < 4) ? bb0[j] : bb1[j - 4];
        float r = fmaxf(acc[j] + bj, 0.0f);
        if (j < 4) o0[j] = r; else o1[j - 4] = r;
        if (col < HW)
          feats[(size_t)n * Cc * HW + row * HW + col] = r + xn[row * HW + col];
      }
      *(v4f*)(&sRT[col * Cc + rb])     = o0;          // reproj transposed [s][c], contiguous
      *(v4f*)(&sRT[col * Cc + rb + 4]) = o1;
    }
  }
  __syncthreads();

  // ---- phase 7: attention_pred = Wl . reproj + bl ----
  if (tid < HW) {
    float acc = bl[0];
    for (int c = 0; c < Cc; ++c) acc += Wl[c] * sRT[tid * Cc + c];
    pred[(size_t)n * HW + tid] = acc;
  }
}

extern "C" void kernel_launch(void* const* d_in, const int* in_sizes, int n_in,
                              void* d_out, int out_size, void* d_ws, size_t ws_size,
                              hipStream_t stream) {
  const float* x  = (const float*)d_in[0];
  const float* Wk = (const float*)d_in[1];
  const float* bk = (const float*)d_in[2];
  const float* Wq = (const float*)d_in[3];
  const float* bq = (const float*)d_in[4];
  const float* Wv = (const float*)d_in[5];
  const float* bv = (const float*)d_in[6];
  const float* Wr = (const float*)d_in[7];
  const float* br = (const float*)d_in[8];
  const float* Wl = (const float*)d_in[9];
  const float* bl = (const float*)d_in[10];

  bf16_t* wk = (bf16_t*)d_ws;
  bf16_t* wq = wk + KCn * Cc;
  bf16_t* wv = wq + KCn * Cc;
  bf16_t* wr = wv + VCn * Cc;

  float* pred  = (float*)d_out;                      // [4096, 1, 49]
  float* feats = pred + (size_t)Nn * HW;             // [4096, 256, 7, 7]

  cvt_weights<<<(VCn * Cc + 255) / 256, 256, 0, stream>>>(Wk, Wq, Wv, Wr, wk, wq, wv, wr);

  // 160 KB dynamic LDS per workgroup (WGP has 320 KB) -> 2 blocks/WGP, 16 waves resident
  attn_head_kernel<<<Nn, 256, 163840, stream>>>(x, bk, bq, bv, br, Wl, bl,
                                                wk, wq, wv, wr, pred, feats);
}
